// MultiHeadAttention_67284957659561
// MI455X (gfx1250) — matrix operand order repaired and checked by reading, not compiled, not run on hardware
//
#include <hip/hip_runtime.h>

// MI455X / gfx1250 multi-head attention.
// Compute-bound (~700 FLOP/byte) -> everything through v_wmma_f32_16x16x32_f16.
// CDNA5 async global->LDS copies (ASYNCcnt) + double-buffered tiles overlap
// data movement with WMMA; flash-attention streaming avoids the 536MB scores.

#define DMODEL 1024
#define HEADS  16
#define DKH    64
#define SEQ    2048
#define BATCH  2
#define NEG_INF -1e9f

typedef __attribute__((ext_vector_type(16))) _Float16 v16h;
typedef __attribute__((ext_vector_type(8)))  _Float16 v8h;
typedef __attribute__((ext_vector_type(8)))  float    v8f;
typedef int b128_t __attribute__((vector_size(16)));   // matches builtin proto

#define AS1 __attribute__((address_space(1)))
#define AS3 __attribute__((address_space(3)))

#if defined(__gfx1250__) && __has_builtin(__builtin_amdgcn_global_load_async_to_lds_b128)
#define USE_ASYNC_LDS 1
#else
#define USE_ASYNC_LDS 0
#endif

// 16-byte global -> LDS copy: async (no VGPR round trip) when available.
static __device__ __forceinline__ void cp16B(const _Float16* g, _Float16* l) {
#if USE_ASYNC_LDS
  __builtin_amdgcn_global_load_async_to_lds_b128((AS1 b128_t*)g, (AS3 b128_t*)l,
                                                 0, 0);
#else
  *(v8h*)l = *(const v8h*)g;
#endif
}

static __device__ __forceinline__ void wait_async() {
#if USE_ASYNC_LDS
#if __has_builtin(__builtin_amdgcn_s_wait_asynccnt)
  __builtin_amdgcn_s_wait_asynccnt(0);
#else
  asm volatile("s_wait_asynccnt 0" ::: "memory");
#endif
#endif
}

// ---- WMMA fragment helpers (ISA 7.12.2 layouts, wave32) -------------------

// A fragment: 16x32 f16, row-major source, leading dim `ld` (elements).
static __device__ __forceinline__ v16h load_frag_a(const _Float16* base, int ld) {
  const int lane    = threadIdx.x & 31;
  const int row     = lane & 15;
  const int halfsel = lane >> 4;
  const _Float16* r = base + row * ld + halfsel * 8;
  v16h a;
#pragma unroll
  for (int i = 0; i < 8; ++i) a[i] = r[i];
#pragma unroll
  for (int i = 0; i < 8; ++i) a[8 + i] = r[16 + i];
  return a;
}

// B fragment: 32x16 f16 (K x N), K-major source, leading dim `ld`.
static __device__ __forceinline__ v16h load_frag_b(const _Float16* base, int ld) {
  const int lane    = threadIdx.x & 31;
  const _Float16* r = base + (lane & 15) + (lane >> 4) * 8 * ld;
  v16h b;
#pragma unroll
  for (int i = 0; i < 8; ++i) b[i] = r[i * ld];
#pragma unroll
  for (int i = 0; i < 8; ++i) b[8 + i] = r[(16 + i) * ld];
  return b;
}

static __device__ __forceinline__ v8f wmma_f16(v16h a, v16h b, v8f c) {
  return __builtin_amdgcn_wmma_f32_16x16x32_f16(false, a, false, b, (short)0, c,
                                                false, false);
}

// ---- fp32 -> fp16 conversion ---------------------------------------------

__global__ void cvt_f32_f16(const float* __restrict__ src,
                            _Float16* __restrict__ dst, int n) {
  int i = blockIdx.x * blockDim.x + threadIdx.x;
  const int stride = gridDim.x * blockDim.x;
  for (; i < n; i += stride) dst[i] = (_Float16)src[i];
}

// ---- tiled GEMM: out = A @ W^T + bias  ------------------------------------
// Block tile 256(M) x 64(N), K-step 32, double-buffered LDS, async prefetch.
// 8 waves, each owns 32x64 -> 8 WMMAs per K-step from 2 A-frags + 4 B-frags.

#define BM 256
#define BN 64
#define KS 32

template <bool SPLIT_HEADS>
__global__ __launch_bounds__(256)
void gemm_xwT(const _Float16* __restrict__ A,   // M x 1024 row-major
              const _Float16* __restrict__ W,   // 1024 x 1024 row-major (N x K)
              const float* __restrict__ bias,   // 1024
              _Float16* __restrict__ outh,      // (B,H,S,DKH) f16 when SPLIT_HEADS
              float* __restrict__ outf) {       // M x 1024 f32 otherwise
  __shared__ _Float16 As[2][BM][40];            // 32 valid K per row (padded)
  __shared__ _Float16 Bt[2][KS][72];            // W tile transposed: [k][n]

  const int tid     = threadIdx.x;
  const int lane    = tid & 31;
  const int wave    = tid >> 5;
  const int m0      = blockIdx.y * BM;
  const int n0      = blockIdx.x * BN;
  const int wm      = wave * 32;
  const int halfsel = lane >> 4;
  const int ncol    = lane & 15;

  v8f acc[2][4] = {};

  auto issue = [&](int st, int k0) {
    // A tile 256x32: thread t copies row t (64B) via 4 async b128
    const _Float16* src = A + (size_t)(m0 + tid) * DMODEL + k0;
#pragma unroll
    for (int i = 0; i < 4; ++i) cp16B(src + i * 8, &As[st][tid][i * 8]);
    // W tile 64x32, transposed into LDS (scatter -> VGPR path)
    const int n  = tid >> 2;
    const int kc = (tid & 3) * 8;
    const _Float16* ws = W + (size_t)(n0 + n) * DMODEL + k0 + kc;
    _Float16 tmp[8];
#pragma unroll
    for (int i = 0; i < 8; ++i) tmp[i] = ws[i];
#pragma unroll
    for (int i = 0; i < 8; ++i) Bt[st][kc + i][n] = tmp[i];
  };

  issue(0, 0);
  for (int it = 0; it < DMODEL / KS; ++it) {
    const int st = it & 1;
    wait_async();            // our async writes into As[st] are done
    __syncthreads();         // everyone done reading As/Bt[st^1] of iter it-1
    if (it + 1 < DMODEL / KS) issue(st ^ 1, (it + 1) * KS);

    const v16h a0 = load_frag_a(&As[st][wm][0], 40);
    const v16h a1 = load_frag_a(&As[st][wm + 16][0], 40);
#pragma unroll
    for (int j = 0; j < 4; ++j) {
      const v16h b = load_frag_b(&Bt[st][0][j * 16], 72);
      acc[0][j] = wmma_f16(a0, b, acc[0][j]);
      acc[1][j] = wmma_f16(a1, b, acc[1][j]);
    }
  }

  // C layout: lane n = ncol, row m = v + 8*halfsel (per 16x16 tile)
#pragma unroll
  for (int i = 0; i < 2; ++i)
#pragma unroll
    for (int j = 0; j < 4; ++j)
#pragma unroll
      for (int v = 0; v < 8; ++v) {
        const int gm = m0 + wm + i * 16 + v + halfsel * 8;
        const int gn = n0 + j * 16 + ncol;
        const float val = acc[i][j][v] + bias[gn];
        if (SPLIT_HEADS) {
          const int b = gm >> 11, s = gm & 2047;
          const int h = gn >> 6,  dk = gn & 63;
          outh[(((size_t)(b * HEADS + h)) * SEQ + s) * DKH + dk] = (_Float16)val;
        } else {
          outf[(size_t)gm * DMODEL + gn] = val;
        }
      }
}

// ---- flash attention: one (b,h), 128 queries/block, 64-key double-buffered -

__global__ __launch_bounds__(256)
void flash_attn(const _Float16* __restrict__ Q,
                const _Float16* __restrict__ K,
                const _Float16* __restrict__ V,
                const int* __restrict__ mask,
                _Float16* __restrict__ Oh) {    // (B,S,DMODEL), head cols packed
  __shared__ _Float16 Kt[2][DKH][72];           // keys transposed: [d][key]
  __shared__ _Float16 Vt[2][64][72];            // [key][d]
  __shared__ _Float16 Ps[8][16][72];            // per-wave P staging [q][key]

  const int tid     = threadIdx.x;
  const int lane    = tid & 31;
  const int wave    = tid >> 5;
  const int halfsel = lane >> 4;
  const int ncol    = lane & 15;

  const int h  = blockIdx.y;
  const int b  = blockIdx.z;
  const int q0 = blockIdx.x * 128;

  const size_t bh = (size_t)(b * HEADS + h);
  const _Float16* Qb = Q + bh * SEQ * DKH;
  const _Float16* Kb = K + bh * SEQ * DKH;
  const _Float16* Vb = V + bh * SEQ * DKH;
  const int* maskb   = mask + b * SEQ;

  // loop-invariant Q fragments: 16 rows x 64 d -> two k-steps
  const _Float16* qrow = Qb + (size_t)(q0 + wave * 16) * DKH;
  const v16h aq0 = load_frag_a(qrow, DKH);
  const v16h aq1 = load_frag_a(qrow + 32, DKH);

  v8f acc_o[4] = {};
  float m_i[8], l_i[8];
#pragma unroll
  for (int v = 0; v < 8; ++v) { m_i[v] = -3.0e38f; l_i[v] = 0.0f; }

  auto issue_kv = [&](int st, int kb) {
    const int key = tid >> 2;
    const int dc  = (tid & 3) * 16;
    // V straight copy: async global->LDS
    const _Float16* vs = Vb + (size_t)(kb + key) * DKH + dc;
    cp16B(vs,     &Vt[st][key][dc]);
    cp16B(vs + 8, &Vt[st][key][dc + 8]);
    // K transposed: VGPR scatter path
    const _Float16* ks = Kb + (size_t)(kb + key) * DKH + dc;
    _Float16 tmp[16];
#pragma unroll
    for (int i = 0; i < 16; ++i) tmp[i] = ks[i];
#pragma unroll
    for (int i = 0; i < 16; ++i) Kt[st][dc + i][key] = tmp[i];
  };

  issue_kv(0, 0);
  for (int it = 0; it < SEQ / 64; ++it) {
    const int st = it & 1;
    const int kb = it * 64;
    wait_async();
    __syncthreads();
    if (it + 1 < SEQ / 64) issue_kv(st ^ 1, kb + 64);

    // scores S = Q K^T : 16 q x 64 keys (4 N-tiles x 2 K-steps)
    v8f sc[4];
#pragma unroll
    for (int nt = 0; nt < 4; ++nt) {
      const v16h bk0 = load_frag_b(&Kt[st][0][nt * 16], 72);
      const v16h bk1 = load_frag_b(&Kt[st][32][nt * 16], 72);
      v8f z = {};
      z = wmma_f16(aq0, bk0, z);
      sc[nt] = wmma_f16(aq1, bk1, z);
      if (maskb[kb + nt * 16 + ncol] == 0) {    // lane owns one key column
#pragma unroll
        for (int v = 0; v < 8; ++v) sc[nt][v] = NEG_INF;
      }
    }

    // online softmax: row m = v + 8*halfsel lives across 16 lanes of a half
    float scale[8];
#pragma unroll
    for (int v = 0; v < 8; ++v) {
      float mx = sc[0][v];
#pragma unroll
      for (int nt = 1; nt < 4; ++nt) mx = fmaxf(mx, sc[nt][v]);
      mx = fmaxf(mx, __shfl_xor(mx, 8, 32));
      mx = fmaxf(mx, __shfl_xor(mx, 4, 32));
      mx = fmaxf(mx, __shfl_xor(mx, 2, 32));
      mx = fmaxf(mx, __shfl_xor(mx, 1, 32));
      const float m_new = fmaxf(m_i[v], mx);
      scale[v] = __expf(m_i[v] - m_new);
      float s = 0.0f;
#pragma unroll
      for (int nt = 0; nt < 4; ++nt) {
        const float p = __expf(sc[nt][v] - m_new);
        sc[nt][v] = p;
        s += p;
      }
      s += __shfl_xor(s, 8, 32);
      s += __shfl_xor(s, 4, 32);
      s += __shfl_xor(s, 2, 32);
      s += __shfl_xor(s, 1, 32);
      l_i[v] = l_i[v] * scale[v] + s;
      m_i[v] = m_new;
    }

#pragma unroll
    for (int dt = 0; dt < 4; ++dt)
#pragma unroll
      for (int v = 0; v < 8; ++v) acc_o[dt][v] *= scale[v];

    // stage P (f16) through LDS to convert C-layout -> A-layout (same wave)
#pragma unroll
    for (int nt = 0; nt < 4; ++nt)
#pragma unroll
      for (int v = 0; v < 8; ++v)
        Ps[wave][v + halfsel * 8][nt * 16 + ncol] = (_Float16)sc[nt][v];
    __builtin_amdgcn_wave_barrier();
    asm volatile("s_wait_dscnt 0" ::: "memory");

    const v16h ap0 = load_frag_a(&Ps[wave][0][0], 72);
    const v16h ap1 = load_frag_a(&Ps[wave][0][32], 72);
#pragma unroll
    for (int dt = 0; dt < 4; ++dt) {
      const v16h bv0 = load_frag_b(&Vt[st][0][dt * 16], 72);
      const v16h bv1 = load_frag_b(&Vt[st][32][dt * 16], 72);
      acc_o[dt] = wmma_f16(ap0, bv0, acc_o[dt]);
      acc_o[dt] = wmma_f16(ap1, bv1, acc_o[dt]);
    }
  }

  // normalize and scatter this head's 64 columns into (B,S,DMODEL)
#pragma unroll
  for (int v = 0; v < 8; ++v) {
    const float inv_l = 1.0f / l_i[v];
    const int sq = q0 + wave * 16 + v + halfsel * 8;
#pragma unroll
    for (int dt = 0; dt < 4; ++dt) {
      const int dc = dt * 16 + ncol;
      Oh[((size_t)b * SEQ + sq) * DMODEL + h * DKH + dc] =
          (_Float16)(acc_o[dt][v] * inv_l);
    }
  }
}

// ---- host orchestration ---------------------------------------------------

extern "C" void kernel_launch(void* const* d_in, const int* in_sizes, int n_in,
                              void* d_out, int out_size, void* d_ws, size_t ws_size,
                              hipStream_t stream) {
  const float* x  = (const float*)d_in[0];
  const int* mask = (const int*)d_in[1];
  const float* Wq = (const float*)d_in[2];
  const float* bq = (const float*)d_in[3];
  const float* Wk = (const float*)d_in[4];
  const float* bk = (const float*)d_in[5];
  const float* Wv = (const float*)d_in[6];
  const float* bv = (const float*)d_in[7];
  const float* Wo = (const float*)d_in[8];
  const float* bo = (const float*)d_in[9];
  float* out = (float*)d_out;

  const size_t M = (size_t)BATCH * SEQ;                 // 4096
  char* p = (char*)d_ws;
  _Float16* xh  = (_Float16*)p; p += M * DMODEL * 2;
  _Float16* wqh = (_Float16*)p; p += (size_t)DMODEL * DMODEL * 2;
  _Float16* wkh = (_Float16*)p; p += (size_t)DMODEL * DMODEL * 2;
  _Float16* wvh = (_Float16*)p; p += (size_t)DMODEL * DMODEL * 2;
  _Float16* woh = (_Float16*)p; p += (size_t)DMODEL * DMODEL * 2;
  _Float16* qh  = (_Float16*)p; p += M * DMODEL * 2;
  _Float16* kh  = (_Float16*)p; p += M * DMODEL * 2;
  _Float16* vh  = (_Float16*)p; p += M * DMODEL * 2;
  _Float16* ah  = (_Float16*)p; p += M * DMODEL * 2;

  cvt_f32_f16<<<dim3(512), dim3(256), 0, stream>>>(x,  xh,  (int)(M * DMODEL));
  cvt_f32_f16<<<dim3(256), dim3(256), 0, stream>>>(Wq, wqh, DMODEL * DMODEL);
  cvt_f32_f16<<<dim3(256), dim3(256), 0, stream>>>(Wk, wkh, DMODEL * DMODEL);
  cvt_f32_f16<<<dim3(256), dim3(256), 0, stream>>>(Wv, wvh, DMODEL * DMODEL);
  cvt_f32_f16<<<dim3(256), dim3(256), 0, stream>>>(Wo, woh, DMODEL * DMODEL);

  const dim3 gg(DMODEL / BN, (unsigned)(M / BM));       // (16, 16)
  gemm_xwT<true><<<gg, 256, 0, stream>>>(xh, wqh, bq, qh, nullptr);
  gemm_xwT<true><<<gg, 256, 0, stream>>>(xh, wkh, bk, kh, nullptr);
  gemm_xwT<true><<<gg, 256, 0, stream>>>(xh, wvh, bv, vh, nullptr);

  flash_attn<<<dim3(SEQ / 128, HEADS, BATCH), 256, 0, stream>>>(qh, kh, vh, mask, ah);

  gemm_xwT<false><<<gg, 256, 0, stream>>>(ah, woh, bo, nullptr, out);
}
